// GrokBaselineModel_15410342658368
// MI455X (gfx1250) — matrix operand order repaired and checked by reading, not compiled, not run on hardware
//
#include <hip/hip_runtime.h>
#include <hip/hip_bf16.h>
#include <math.h>

// Problem constants (match reference)
#define BB   2
#define SS   2048
#define DD   1024
#define HH   16
#define DHD  64
#define LL   2
#define FF   2736
#define FP   2752          // F padded to multiple of 64 for WMMA/TDM staging
#define MM   (BB*SS)       // 4096 rows
#define EPSV 1e-5f
#define CAPV 30.0f
#define NEGV -10000.0f

typedef __attribute__((ext_vector_type(16))) _Float16     v16h;
typedef __attribute__((ext_vector_type(8)))  float        v8f;
typedef __attribute__((ext_vector_type(4)))  unsigned int v4u;
typedef __attribute__((ext_vector_type(8)))  int          v8i;
typedef __attribute__((ext_vector_type(4)))  int          v4i;

union Frag16 { v16h v; float4 q[2]; _Float16 h[16]; };

// ---- WMMA fragment loaders (layouts per CDNA5 ISA §7.12.2) ----

// A-matrix 16x32 f16: lane<16 -> M=lane, K in {0..7,16..23}; lane>=16 -> K in {8..15,24..31}
__device__ __forceinline__ v16h load_a_frag(const _Float16* __restrict__ A, int lda,
                                            int row0, int k0, int lane) {
  Frag16 f;
  int m  = lane & 15;
  int ks = (lane >> 4) << 3;               // 0 or 8 halves
  const _Float16* p = A + (size_t)(row0 + m) * lda + k0 + ks;
  f.q[0] = *reinterpret_cast<const float4*>(p);
  f.q[1] = *reinterpret_cast<const float4*>(p + 16);
  return f.v;
}

// B-matrix 32x16 f16: lane holds column N=lane&15; K in {0..7,16..23} (lane<16) or {8..15,24..31} (lane>=16)
__device__ __forceinline__ v16h load_b_frag(const _Float16* __restrict__ Bp, int ldb,
                                            int col0, int k0, int lane) {
  Frag16 f;
  int n  = lane & 15;
  int ks = (lane >> 4) << 3;               // 0 or 8 halves
  const _Float16* p = Bp + (size_t)(col0 + n) * ldb + k0 + ks;
  f.q[0] = *reinterpret_cast<const float4*>(p);
  f.q[1] = *reinterpret_cast<const float4*>(p + 16);
  return f.v;
}

__device__ __forceinline__ v8f wmma_f16(v16h a, v16h b, v8f c) {
  return __builtin_amdgcn_wmma_f32_16x16x32_f16(false, a, false, b, (short)0, c, false, false);
}

__device__ __forceinline__ void store_tile(float* __restrict__ C, int ldc,
                                           int row0, int col0, v8f acc, int lane) {
  int mrow = row0 + ((lane >> 4) << 3);
  int ncol = col0 + (lane & 15);
  #pragma unroll
  for (int r = 0; r < 8; ++r)
    C[(size_t)(mrow + r) * ldc + ncol] = acc[r];
}

// The TDM builtin writes LDS without a pointer the compiler can see, so LDS
// loads after it would otherwise fold to undef ("no stores to this global").
// Escape the LDS base into an asm with a memory clobber: the asm may write
// through it, so subsequent ds_loads must actually read memory; provenance of
// the loads is unchanged, so they remain DS ops.
__device__ __forceinline__ void lds_publish(const void* p) {
  asm volatile("" :: "s"(p) : "memory");
}

// ---- Tensor Data Mover: 2-D tile (rows x 64 f16) global -> LDS ----
// D# layout per CDNA5 ISA (group0) / §8.4 (group1).
__device__ __forceinline__ void tdm_load_2d(unsigned lds_off_bytes,
                                            const _Float16* gptr,
                                            unsigned tile_d0, unsigned tile_d1,
                                            unsigned tensor_d0, unsigned tensor_d1,
                                            unsigned stride_elems)
{
  unsigned long long ga = (unsigned long long)(size_t)gptr;
  v4u g0;
  g0[0] = 1u;                                    // count=1, user descriptor
  g0[1] = lds_off_bytes;                         // lds_addr (bytes)
  g0[2] = (unsigned)(ga & 0xffffffffu);          // global_addr[31:0]
  g0[3] = (unsigned)(ga >> 32) | (2u << 30);     // global_addr[56:32] | type=2
  v8i g1;
  g1[0] = (int)(1u << 16);                       // data_size=1 (2 bytes/elem)
  g1[1] = (int)((tensor_d0 & 0xffffu) << 16);    // tensor_dim0[15:0] @ bits63:48
  g1[2] = (int)((tensor_d0 >> 16) | ((tensor_d1 & 0xffffu) << 16));
  g1[3] = (int)((tensor_d1 >> 16) | (tile_d0 << 16));   // tile_dim0 @127:112
  g1[4] = (int)(tile_d1 & 0xffffu);              // tile_dim1; tile_dim2=0
  g1[5] = (int)stride_elems;                     // tensor_dim0_stride[31:0]
  g1[6] = 0;                                     // stride[47:32]=0, dim1_stride lo=0
  g1[7] = 0;
  v4i z4 = {0, 0, 0, 0};
#if __clang_major__ >= 23
  v8i z8 = {0, 0, 0, 0, 0, 0, 0, 0};
  __builtin_amdgcn_tensor_load_to_lds(g0, g1, z4, z4, z8, 0);
#else
  __builtin_amdgcn_tensor_load_to_lds(g0, g1, z4, z4, 0);
#endif
}

// ---- GEMM: C[M,N] f32 = A[M,K] f16 @ W[N,K]^T f16, TDM-staged LDS, 128x64 macro ----
// 8 waves: wave (wm,wn) in 4x2, each computes 32x32 (2x2 WMMA tiles).
// Requires M%128==0, N%64==0, K%64==0.
__global__ __launch_bounds__(256) void gemm_f16_wmma_tdm(
    const _Float16* __restrict__ A, const _Float16* __restrict__ W,
    float* __restrict__ C, int Mm, int Nn, int Kk, int lda, int ldb, int ldc)
{
  __shared__ _Float16 smem[24576];               // A:128x64 (x2) + B:64x64 (x2) = 48 KB
  const unsigned AOFF0 = 0, BOFF0 = 8192, AOFF1 = 12288, BOFF1 = 16384;  // halves
  int wave = threadIdx.x >> 5, lane = threadIdx.x & 31;
  int wm = wave >> 1, wn = wave & 1;
  int tiles_n = Nn / 64;
  int bm = (blockIdx.x / tiles_n) * 128;
  int bn = (blockIdx.x % tiles_n) * 64;
  const _Float16* Ab = A + (size_t)bm * lda;
  const _Float16* Wb = W + (size_t)bn * ldb;
  int nstages = Kk / 64;

  if (wave == 0) {
    tdm_load_2d(AOFF0 * 2, Ab, 64, 128, (unsigned)Kk, (unsigned)Mm, (unsigned)lda);
    tdm_load_2d(BOFF0 * 2, Wb, 64, 64,  (unsigned)Kk, (unsigned)Nn, (unsigned)ldb);
    __builtin_amdgcn_s_wait_tensorcnt(0);
  }
  lds_publish(smem);
  __syncthreads();

  v8f z = (v8f){0.f,0.f,0.f,0.f,0.f,0.f,0.f,0.f};
  v8f acc00 = z, acc01 = z, acc10 = z, acc11 = z;

  for (int s = 0; s < nstages; ++s) {
    int buf = s & 1;
    if (wave == 0 && s + 1 < nstages) {
      int k1 = (s + 1) * 64;
      tdm_load_2d((buf ? AOFF0 : AOFF1) * 2, Ab + k1, 64, 128, (unsigned)Kk, (unsigned)Mm, (unsigned)lda);
      tdm_load_2d((buf ? BOFF0 : BOFF1) * 2, Wb + k1, 64, 64,  (unsigned)Kk, (unsigned)Nn, (unsigned)ldb);
    }
    const _Float16* As = smem + (buf ? AOFF1 : AOFF0);
    const _Float16* Bs = smem + (buf ? BOFF1 : BOFF0);
    #pragma unroll
    for (int kk = 0; kk < 64; kk += 32) {
      v16h a0 = load_a_frag(As, 64, wm * 32,      kk, lane);
      v16h a1 = load_a_frag(As, 64, wm * 32 + 16, kk, lane);
      v16h b0 = load_b_frag(Bs, 64, wn * 32,      kk, lane);
      v16h b1 = load_b_frag(Bs, 64, wn * 32 + 16, kk, lane);
      acc00 = wmma_f16(a0, b0, acc00);
      acc01 = wmma_f16(a0, b1, acc01);
      acc10 = wmma_f16(a1, b0, acc10);
      acc11 = wmma_f16(a1, b1, acc11);
    }
    if (wave == 0 && s + 1 < nstages) __builtin_amdgcn_s_wait_tensorcnt(0);
    lds_publish(smem);
    __syncthreads();
  }

  store_tile(C, ldc, bm + wm*32,      bn + wn*32,      acc00, lane);
  store_tile(C, ldc, bm + wm*32,      bn + wn*32 + 16, acc01, lane);
  store_tile(C, ldc, bm + wm*32 + 16, bn + wn*32,      acc10, lane);
  store_tile(C, ldc, bm + wm*32 + 16, bn + wn*32 + 16, acc11, lane);
}

// ---- RMSNorm -> f16 ----
__global__ __launch_bounds__(256) void rmsnorm_f16_kernel(
    const float* __restrict__ x, const float* __restrict__ sc, _Float16* __restrict__ out)
{
  __shared__ float red[256];
  int row = blockIdx.x, tid = threadIdx.x;
  const float* xr = x + (size_t)row * DD;
  float ss = 0.f;
  #pragma unroll
  for (int i = 0; i < DD/256; ++i) { float v = xr[tid + i*256]; ss += v*v; }
  red[tid] = ss; __syncthreads();
  for (int s = 128; s > 0; s >>= 1) { if (tid < s) red[tid] += red[tid+s]; __syncthreads(); }
  float inv = rsqrtf(red[0] / (float)DD + EPSV);
  _Float16* orow = out + (size_t)row * DD;
  #pragma unroll
  for (int i = 0; i < DD/256; ++i) {
    int d = tid + i*256;
    orow[d] = (_Float16)(xr[d] * inv * sc[d]);
  }
}

// ---- h += rmsnorm(y, sc); optional pad handling ----
__global__ __launch_bounds__(256) void add_rmsnorm_kernel(
    float* __restrict__ h, const float* __restrict__ y, const float* __restrict__ sc,
    const unsigned char* __restrict__ pad, int zero_y_if_pad, int zero_out_if_pad)
{
  __shared__ float red[256];
  int row = blockIdx.x, tid = threadIdx.x;
  bool p = pad[row] != 0;
  bool zy = p && zero_y_if_pad;
  const float* yr = y + (size_t)row * DD;
  float* hr = h + (size_t)row * DD;
  float ss = 0.f;
  #pragma unroll
  for (int i = 0; i < DD/256; ++i) {
    float v = zy ? 0.f : yr[tid + i*256];
    ss += v*v;
  }
  red[tid] = ss; __syncthreads();
  for (int s = 128; s > 0; s >>= 1) { if (tid < s) red[tid] += red[tid+s]; __syncthreads(); }
  float inv = rsqrtf(red[0] / (float)DD + EPSV);
  #pragma unroll
  for (int i = 0; i < DD/256; ++i) {
    int d = tid + i*256;
    float v = zy ? 0.f : yr[d];
    float hn = hr[d] + v * inv * sc[d];
    if (p && zero_out_if_pad) hn = 0.f;
    hr[d] = hn;
  }
}

// ---- RoPE on Q,K + convert; K -> [B,H,S,DH] f16, V -> [B,H,DH,S] f16 (transposed) ----
__global__ __launch_bounds__(256) void rope_cvt_kernel(
    const float* __restrict__ qf, const float* __restrict__ kf, const float* __restrict__ vf,
    _Float16* __restrict__ qh, _Float16* __restrict__ kh, _Float16* __restrict__ vt)
{
  int idx = blockIdx.x * 256 + threadIdx.x;       // < M*D = 2^22
  int d  = idx & 63;
  int hh = (idx >> 6) & 15;
  int s  = (idx >> 10) & 2047;
  int b  = idx >> 21;
  size_t rowbase = (size_t)(b*SS + s) * DD + hh*DHD;
  int j = d >> 1;
  float freq = powf(10000.f, -((float)(2*j)) / 64.f);
  float ph = (float)s * freq;
  float c = cosf(ph), sn = sinf(ph);
  int d2 = (d < 32) ? d + 32 : d - 32;
  float qv = qf[rowbase + d], kv = kf[rowbase + d];
  float qo = qf[rowbase + d2], ko = kf[rowbase + d2];
  float qr = (d < 32) ? -qo : qo;
  float kr = (d < 32) ? -ko : ko;
  size_t dsth = ((size_t)(b*HH + hh) * SS + s) * DHD + d;
  qh[dsth] = (_Float16)(qv*c + qr*sn);
  kh[dsth] = (_Float16)(kv*c + kr*sn);
  vt[((size_t)(b*HH + hh) * DHD + d) * SS + s] = (_Float16)vf[rowbase + d];
}

// ---- Flash attention: one wave per (b, h, 16-query tile), 32 keys per step ----
__global__ __launch_bounds__(32) void flash_attn_kernel(
    const _Float16* __restrict__ qh, const _Float16* __restrict__ kh,
    const _Float16* __restrict__ vt,
    const unsigned char* __restrict__ allowed, const unsigned char* __restrict__ pad,
    _Float16* __restrict__ atth)
{
  __shared__ _Float16 Pt[16][32];                 // P transpose staging (1 KB)
  int lane = threadIdx.x;
  int idx  = blockIdx.x;
  int qt = idx & ((SS/16) - 1);
  int hh = (idx >> 7) & (HH - 1);
  int b  = idx >> 11;
  int q0 = qt << 4;
  const _Float16* qbase = qh + (size_t)(b*HH + hh) * SS * DHD;
  const _Float16* kbase = kh + (size_t)(b*HH + hh) * SS * DHD;
  const _Float16* vbase = vt + (size_t)(b*HH + hh) * DHD * SS;

  v16h aq0 = load_a_frag(qbase, DHD, q0, 0,  lane);
  v16h aq1 = load_a_frag(qbase, DHD, q0, 32, lane);

  v8f accO[4];
  float mi[8], li[8];
  #pragma unroll
  for (int nt = 0; nt < 4; ++nt) accO[nt] = (v8f){0.f,0.f,0.f,0.f,0.f,0.f,0.f,0.f};
  #pragma unroll
  for (int r = 0; r < 8; ++r) { mi[r] = -1e30f; li[r] = 0.f; }

  int nhalf = lane >> 4;
  int ncol  = lane & 15;
  const float SC = 0.125f / CAPV;                 // (1/sqrt(DH)) / CAP

  for (int kt = 0; kt < SS/32; ++kt) {
    int kb = kt * 32;
    if (kb + 32 < SS) {                           // prefetch next key/value tiles
      __builtin_prefetch(kbase + (size_t)(kb + 32) * DHD, 0, 1);
      __builtin_prefetch(vbase + (size_t)(kb + 32), 0, 1);
    }
    v8f s0 = (v8f){0.f,0.f,0.f,0.f,0.f,0.f,0.f,0.f};
    v8f s1 = s0;
    s0 = wmma_f16(aq0, load_b_frag(kbase, DHD, kb,      0,  lane), s0);
    s0 = wmma_f16(aq1, load_b_frag(kbase, DHD, kb,      32, lane), s0);
    s1 = wmma_f16(aq0, load_b_frag(kbase, DHD, kb + 16, 0,  lane), s1);
    s1 = wmma_f16(aq1, load_b_frag(kbase, DHD, kb + 16, 32, lane), s1);

    int key0 = kb + ncol, key1 = kb + 16 + ncol;
    bool kp0 = pad[b*SS + key0] == 0;
    bool kp1 = pad[b*SS + key1] == 0;
    float p0[8], p1[8];
    #pragma unroll
    for (int r = 0; r < 8; ++r) {
      int q = q0 + r + (nhalf << 3);
      float l0 = CAPV * tanhf(s0[r] * SC);
      float l1 = CAPV * tanhf(s1[r] * SC);
      l0 = (allowed[(size_t)q*SS + key0] && kp0) ? l0 : NEGV;
      l1 = (allowed[(size_t)q*SS + key1] && kp1) ? l1 : NEGV;
      float mx = fmaxf(l0, l1);
      #pragma unroll
      for (int o = 1; o < 16; o <<= 1) mx = fmaxf(mx, __shfl_xor(mx, o, 32));
      float mnew  = fmaxf(mi[r], mx);
      float alpha = expf(mi[r] - mnew);
      mi[r] = mnew;
      float e0 = expf(l0 - mnew), e1 = expf(l1 - mnew);
      float rs = e0 + e1;
      #pragma unroll
      for (int o = 1; o < 16; o <<= 1) rs += __shfl_xor(rs, o, 32);
      li[r] = li[r] * alpha + rs;
      p0[r] = e0; p1[r] = e1;
      accO[0][r] *= alpha; accO[1][r] *= alpha;
      accO[2][r] *= alpha; accO[3][r] *= alpha;
    }
    __syncthreads();
    #pragma unroll
    for (int r = 0; r < 8; ++r) {
      int m = r + (nhalf << 3);
      Pt[m][ncol]      = (_Float16)p0[r];
      Pt[m][16 + ncol] = (_Float16)p1[r];
    }
    __syncthreads();
    v16h pa = load_a_frag(&Pt[0][0], 32, 0, 0, lane);
    #pragma unroll
    for (int nt = 0; nt < 4; ++nt) {
      v16h bv = load_b_frag(vbase, SS, nt*16, kb, lane);
      accO[nt] = wmma_f16(pa, bv, accO[nt]);
    }
    __syncthreads();
  }
  #pragma unroll
  for (int nt = 0; nt < 4; ++nt) {
    #pragma unroll
    for (int r = 0; r < 8; ++r) {
      int q = q0 + r + (nhalf << 3);
      float o = accO[nt][r] / fmaxf(li[r], 1e-20f);
      atth[(size_t)(b*SS + q) * DD + hh*DHD + nt*16 + ncol] = (_Float16)o;
    }
  }
}

// ---- act[m, fpad] = gelu_exact(gate)*val (f16, zero tail); gate/val use ldc=FP ----
__global__ __launch_bounds__(256) void geluval_kernel(
    const float* __restrict__ gate, const float* __restrict__ val, _Float16* __restrict__ act)
{
  size_t idx = (size_t)blockIdx.x * 256 + threadIdx.x;
  if (idx >= (size_t)MM * FP) return;
  int f = (int)(idx % FP);
  _Float16 o = (_Float16)0.f;
  if (f < FF) {
    float g = gate[idx];
    float v = val[idx];
    float ge = 0.5f * g * (1.f + erff(g * 0.70710678118f));
    o = (_Float16)(ge * v);
  }
  act[idx] = o;
}

// ---- f32 -> f16 weight convert with row+column zero pad ----
__global__ __launch_bounds__(256) void cvt_pad_kernel(
    const float* __restrict__ src, _Float16* __restrict__ dst,
    int srows, int scols, int drows, int dcols)
{
  size_t idx = (size_t)blockIdx.x * 256 + threadIdx.x;
  if (idx >= (size_t)drows * dcols) return;
  int c = (int)(idx % dcols);
  size_t r = idx / dcols;
  dst[idx] = (r < (size_t)srows && c < scols) ? (_Float16)src[r * (size_t)scols + c]
                                              : (_Float16)0.f;
}

// ---------------------------------------------------------------------------

static inline size_t alignUp(size_t x) { return (x + 255) & ~(size_t)255; }

extern "C" void kernel_launch(void* const* d_in, const int* in_sizes, int n_in,
                              void* d_out, int out_size, void* d_ws, size_t ws_size,
                              hipStream_t stream) {
  const float*         hid     = (const float*)d_in[0];
  const unsigned char* allowed = (const unsigned char*)d_in[1];
  const unsigned char* pad     = (const unsigned char*)d_in[2];
  const float*         rms     = (const float*)d_in[3];
  const float*         Wq      = (const float*)d_in[4];
  const float*         Wk      = (const float*)d_in[5];
  const float*         Wv      = (const float*)d_in[6];
  const float*         Wo      = (const float*)d_in[7];
  const float*         Wg      = (const float*)d_in[8];
  const float*         Wvl     = (const float*)d_in[9];
  const float*         Wout    = (const float*)d_in[10];

  size_t off = 0;
  char* ws = (char*)d_ws;
  auto alloc = [&](size_t bytes) { void* p = ws + off; off += alignUp(bytes); return p; };

  float*     h    = (float*)    alloc((size_t)MM*DD*4);
  _Float16*  xh   = (_Float16*) alloc((size_t)MM*DD*2);
  float*     qf   = (float*)    alloc((size_t)MM*DD*4);   // also reused as y32
  float*     kf   = (float*)    alloc((size_t)MM*DD*4);
  float*     vf   = (float*)    alloc((size_t)MM*DD*4);
  _Float16*  qh   = (_Float16*) alloc((size_t)MM*DD*2);
  _Float16*  kh   = (_Float16*) alloc((size_t)MM*DD*2);
  _Float16*  vt   = (_Float16*) alloc((size_t)MM*DD*2);
  _Float16*  atth = (_Float16*) alloc((size_t)MM*DD*2);
  float*     gate = (float*)    alloc((size_t)MM*FP*4);
  float*     val  = (float*)    alloc((size_t)MM*FP*4);
  _Float16*  acth = (_Float16*) alloc((size_t)MM*FP*2);
  _Float16*  wqh  = (_Float16*) alloc((size_t)DD*DD*2);
  _Float16*  wkh  = (_Float16*) alloc((size_t)DD*DD*2);
  _Float16*  wvh  = (_Float16*) alloc((size_t)DD*DD*2);
  _Float16*  woh  = (_Float16*) alloc((size_t)DD*DD*2);
  _Float16*  wgh  = (_Float16*) alloc((size_t)FP*DD*2);   // rows padded FF->FP
  _Float16*  wvlh = (_Float16*) alloc((size_t)FP*DD*2);
  _Float16*  wouth= (_Float16*) alloc((size_t)DD*FP*2);   // cols padded FF->FP
  float*     y32  = qf;                                   // free after rope

  hipMemcpyAsync(h, hid, (size_t)MM*DD*4, hipMemcpyDeviceToDevice, stream);

  auto grid1 = [](size_t n) { return (unsigned)((n + 255) / 256); };
  const unsigned gQKVO = (MM/128) * (DD/64);   // 512 workgroups
  const unsigned gFFN  = (MM/128) * (FP/64);   // 1376 workgroups

  for (int l = 0; l < LL; ++l) {
    const float* sc = rms + (size_t)l * 4 * DD;

    // convert this layer's weights to f16 (padded where needed)
    cvt_pad_kernel<<<grid1((size_t)DD*DD), 256, 0, stream>>>(Wq  + (size_t)l*DD*DD, wqh,  DD, DD, DD, DD);
    cvt_pad_kernel<<<grid1((size_t)DD*DD), 256, 0, stream>>>(Wk  + (size_t)l*DD*DD, wkh,  DD, DD, DD, DD);
    cvt_pad_kernel<<<grid1((size_t)DD*DD), 256, 0, stream>>>(Wv  + (size_t)l*DD*DD, wvh,  DD, DD, DD, DD);
    cvt_pad_kernel<<<grid1((size_t)DD*DD), 256, 0, stream>>>(Wo  + (size_t)l*DD*DD, woh,  DD, DD, DD, DD);
    cvt_pad_kernel<<<grid1((size_t)FP*DD), 256, 0, stream>>>(Wg  + (size_t)l*FF*DD, wgh,  FF, DD, FP, DD);
    cvt_pad_kernel<<<grid1((size_t)FP*DD), 256, 0, stream>>>(Wvl + (size_t)l*FF*DD, wvlh, FF, DD, FP, DD);
    cvt_pad_kernel<<<grid1((size_t)DD*FP), 256, 0, stream>>>(Wout+ (size_t)l*DD*FF, wouth,DD, FF, DD, FP);

    // attention block
    rmsnorm_f16_kernel<<<MM, 256, 0, stream>>>(h, sc + 0*DD, xh);
    gemm_f16_wmma_tdm<<<gQKVO, 256, 0, stream>>>(xh, wqh, qf, MM, DD, DD, DD, DD, DD);
    gemm_f16_wmma_tdm<<<gQKVO, 256, 0, stream>>>(xh, wkh, kf, MM, DD, DD, DD, DD, DD);
    gemm_f16_wmma_tdm<<<gQKVO, 256, 0, stream>>>(xh, wvh, vf, MM, DD, DD, DD, DD, DD);
    rope_cvt_kernel<<<grid1((size_t)MM*DD), 256, 0, stream>>>(qf, kf, vf, qh, kh, vt);
    flash_attn_kernel<<<BB*HH*(SS/16), 32, 0, stream>>>(qh, kh, vt, allowed, pad, atth);
    gemm_f16_wmma_tdm<<<gQKVO, 256, 0, stream>>>(atth, woh, y32, MM, DD, DD, DD, DD, DD);
    add_rmsnorm_kernel<<<MM, 256, 0, stream>>>(h, y32, sc + 1*DD, pad, 1, 0);

    // FFN block
    rmsnorm_f16_kernel<<<MM, 256, 0, stream>>>(h, sc + 2*DD, xh);
    gemm_f16_wmma_tdm<<<gFFN, 256, 0, stream>>>(xh, wgh,  gate, MM, FP, DD, DD, DD, FP);
    gemm_f16_wmma_tdm<<<gFFN, 256, 0, stream>>>(xh, wvlh, val,  MM, FP, DD, DD, DD, FP);
    geluval_kernel<<<grid1((size_t)MM*FP), 256, 0, stream>>>(gate, val, acth);
    gemm_f16_wmma_tdm<<<gQKVO, 256, 0, stream>>>(acth, wouth, y32, MM, DD, FP, FP, FP, DD);
    add_rmsnorm_kernel<<<MM, 256, 0, stream>>>(h, y32, sc + 3*DD, pad, 0, 1);
  }

  hipMemcpyAsync(d_out, h, (size_t)MM*DD*4, hipMemcpyDeviceToDevice, stream);
}
